// TransformerEncoderKANRegressor_8014408974697
// MI455X (gfx1250) — matrix operand order repaired and checked by reading, not compiled, not run on hardware
//
#include <hip/hip_runtime.h>
#include <math.h>

// ---------------- types / WMMA helpers ----------------
typedef __attribute__((ext_vector_type(16))) __bf16 bf16x16;
typedef __attribute__((ext_vector_type(8)))  float  f32x8;

union BFrag { bf16x16 v; uint4 u[2]; };

__device__ __forceinline__ f32x8 wmma_bf16(const BFrag& a, const BFrag& b, f32x8 c) {
  return __builtin_amdgcn_wmma_f32_16x16x32_bf16(false, a.v, false, b.v, (short)0, c, false, false);
}

__device__ __forceinline__ float wave_sum(float v) {
  #pragma unroll
  for (int m = 16; m >= 1; m >>= 1) v += __shfl_xor(v, m, 32);
  return v;
}
__device__ __forceinline__ float wave_max(float v) {
  #pragma unroll
  for (int m = 16; m >= 1; m >>= 1) v = fmaxf(v, __shfl_xor(v, m, 32));
  return v;
}

// ---------------- async global->LDS staging (CDNA5 path, with fallback) ----------------
#if defined(__has_builtin)
# if __has_builtin(__builtin_amdgcn_global_load_async_to_lds_b128)
#  define HAVE_ASYNC_LDS 1
# endif
#endif

#ifdef HAVE_ASYNC_LDS
typedef int v4i_ __attribute__((vector_size(16)));
__device__ __forceinline__ void cp_b128(__bf16* lds, const __bf16* g) {
  void* gv = (void*)const_cast<__bf16*>(g);
  void* lv = (void*)lds;
  __builtin_amdgcn_global_load_async_to_lds_b128(
      (__attribute__((address_space(1))) v4i_*)gv,
      (__attribute__((address_space(3))) v4i_*)lv,
      0, 0);
}
__device__ __forceinline__ void async_wait() {
  asm volatile("s_wait_asynccnt 0" ::: "memory");
}
#else
__device__ __forceinline__ void cp_b128(__bf16* lds, const __bf16* g) {
  *(uint4*)lds = *(const uint4*)g;
}
__device__ __forceinline__ void async_wait() {}
#endif

// ---------------- model constants ----------------
static constexpr int Bv = 64, Fv = 256, Dv = 256, Hv = 512, Lv = 3, NHv = 8, OUTv = 2;
static constexpr int Sq = 256;            // sequence length (== Fv)
static constexpr int HD = 32;             // head dim
static constexpr int NTOK = Bv * Fv;      // 16384 tokens
static constexpr int CB = 16;             // attention batch chunk
static constexpr int CN = 4096;           // KAN row chunk

// ---------------- elementwise kernels ----------------
__global__ void k_f32_to_bf16(const float* __restrict__ s, __bf16* __restrict__ d, int n) {
  int i = blockIdx.x * blockDim.x + threadIdx.x;
  if (i < n) d[i] = (__bf16)s[i];
}

__global__ void k_embed(const float* __restrict__ x, const float* __restrict__ vw,
                        const float* __restrict__ vb, const float* __restrict__ emb,
                        float* __restrict__ h) {
  int i = blockIdx.x * blockDim.x + threadIdx.x;     // over NTOK*Dv
  if (i >= NTOK * Dv) return;
  int d = i & (Dv - 1);
  int t = i >> 8;                                    // token = b*Fv+f
  int f = t & (Fv - 1);
  h[i] = x[t] * vw[d] + vb[d] + emb[f * Dv + d];
}

// Fold KAN weights: Wt[o, i*9+0]=sb[i,o]; Wt[o, i*9+1+g]=ssp[i,o]*coef[i,o,g]
__global__ void k_kan_weight(const float* __restrict__ coef, const float* __restrict__ sb,
                             const float* __restrict__ ssp, __bf16* __restrict__ Wt,
                             int I, int O) {
  int i = blockIdx.x * blockDim.x + threadIdx.x;
  if (i >= I * O) return;
  int o = i % O;
  int ii = i / O;
  float s = ssp[i];
  __bf16* w = Wt + (size_t)o * (I * 9) + ii * 9;
  w[0] = (__bf16)sb[i];
  const float* c = coef + (size_t)i * 8;
  #pragma unroll
  for (int g = 0; g < 8; ++g) w[1 + g] = (__bf16)(s * c[g]);
}

// phi[n, i*9 + t]: slot0 = silu(x), slots 1..8 = cubic B-spline basis (G=5,K=3 uniform grid)
__global__ void k_phi(const float* __restrict__ X, __bf16* __restrict__ Phi, int total) {
  int i = blockIdx.x * blockDim.x + threadIdx.x;
  if (i >= total) return;
  float x = X[i];
  float sig = 1.0f / (1.0f + expf(-x));
  __bf16* w = Phi + (size_t)i * 9;
  w[0] = (__bf16)(x * sig);
  float bb[11];
  #pragma unroll
  for (int j = 0; j < 11; ++j) {
    float tj = 0.4f * j - 2.2f;
    bb[j] = (x >= tj && x < tj + 0.4f) ? 1.0f : 0.0f;
  }
  #pragma unroll
  for (int p = 1; p <= 3; ++p) {
    float invd = 1.0f / (0.4f * p);
    #pragma unroll
    for (int j = 0; j <= 10 - p; ++j) {
      float tj = 0.4f * j - 2.2f;
      bb[j] = (x - tj) * invd * bb[j] + ((tj + 0.4f * (p + 1)) - x) * invd * bb[j + 1];
    }
  }
  #pragma unroll
  for (int g = 0; g < 8; ++g) w[1 + g] = (__bf16)bb[g];
}

// Residual + LayerNorm over Dv=256 columns, one wave per row, in-place into h.
__global__ void k_resid_ln(float* __restrict__ h, const float* __restrict__ a,
                           const float* __restrict__ w, const float* __restrict__ bvec,
                           int rows) {
  int row  = (blockIdx.x * blockDim.x + threadIdx.x) >> 5;
  int lane = threadIdx.x & 31;
  if (row >= rows) return;
  float* hp = h + (size_t)row * Dv + lane * 8;
  float4 v0 = *(float4*)hp, v1 = *(float4*)(hp + 4);
  float x[8] = {v0.x, v0.y, v0.z, v0.w, v1.x, v1.y, v1.z, v1.w};
  if (a) {
    const float* ap = a + (size_t)row * Dv + lane * 8;
    float4 a0 = *(const float4*)ap, a1 = *(const float4*)(ap + 4);
    x[0] += a0.x; x[1] += a0.y; x[2] += a0.z; x[3] += a0.w;
    x[4] += a1.x; x[5] += a1.y; x[6] += a1.z; x[7] += a1.w;
  }
  float s = 0;
  #pragma unroll
  for (int j = 0; j < 8; ++j) s += x[j];
  float mean = wave_sum(s) * (1.0f / Dv);
  float vs = 0;
  #pragma unroll
  for (int j = 0; j < 8; ++j) { float d = x[j] - mean; vs += d * d; }
  float inv = rsqrtf(wave_sum(vs) * (1.0f / Dv) + 1e-5f);
  #pragma unroll
  for (int j = 0; j < 8; ++j) {
    int col = lane * 8 + j;
    x[j] = (x[j] - mean) * inv * w[col] + bvec[col];
  }
  *(float4*)hp       = make_float4(x[0], x[1], x[2], x[3]);
  *(float4*)(hp + 4) = make_float4(x[4], x[5], x[6], x[7]);
}

__global__ void k_softmax(const float* __restrict__ sc, __bf16* __restrict__ pr, int rows) {
  int row  = (blockIdx.x * blockDim.x + threadIdx.x) >> 5;
  int lane = threadIdx.x & 31;
  if (row >= rows) return;
  const float* p = sc + (size_t)row * Sq + lane * 8;
  float4 v0 = *(const float4*)p, v1 = *(const float4*)(p + 4);
  float x[8] = {v0.x, v0.y, v0.z, v0.w, v1.x, v1.y, v1.z, v1.w};
  float mx = x[0];
  #pragma unroll
  for (int j = 1; j < 8; ++j) mx = fmaxf(mx, x[j]);
  mx = wave_max(mx);
  float s = 0;
  #pragma unroll
  for (int j = 0; j < 8; ++j) { x[j] = expf(x[j] - mx); s += x[j]; }
  float inv = 1.0f / wave_sum(s);
  union { __bf16 b[8]; uint4 q; } ob;
  #pragma unroll
  for (int j = 0; j < 8; ++j) ob.b[j] = (__bf16)(x[j] * inv);
  *(uint4*)(pr + (size_t)row * Sq + lane * 8) = ob.q;
}

__global__ void k_vtrans(const __bf16* __restrict__ qkv, __bf16* __restrict__ vT, int b0) {
  int i = blockIdx.x * blockDim.x + threadIdx.x;   // over CB*8*32*256
  if (i >= CB * NHv * HD * Sq) return;
  int sk = i & 255, d = (i >> 8) & 31, hh = (i >> 13) & 7, bc = i >> 16;
  int b = b0 + bc;
  vT[i] = qkv[(size_t)(b * Sq + sk) * 768 + 512 + hh * 32 + d];
}

__global__ void k_pool(const float* __restrict__ h, float* __restrict__ p) {
  int i = blockIdx.x * blockDim.x + threadIdx.x;   // over Bv*Dv
  if (i >= Bv * Dv) return;
  int d = i & 255, b = i >> 8;
  float s = 0;
  for (int t = 0; t < Sq; ++t) s += h[(size_t)(b * Sq + t) * Dv + d];
  p[i] = s * (1.0f / Sq);
}

__global__ void k_reg(const float* __restrict__ k2, const float* __restrict__ w,
                      const float* __restrict__ bvec, float* __restrict__ out) {
  int i = blockIdx.x * blockDim.x + threadIdx.x;
  if (i >= Bv * OUTv) return;
  int o = i & 1, b = i >> 1;
  float s = bvec[o];
  for (int d = 0; d < Hv; ++d) s += k2[(size_t)b * Hv + d] * w[(size_t)o * Hv + d];
  out[i] = s;
}

// ---------------- direct WMMA GEMM (small M; heads) ----------------
// C[M,N] = A[M,K](bf16) @ Bt[N,K](bf16)^T (+bias). M%16==0, N%64==0, K%32==0.
__global__ void k_gemm(const __bf16* __restrict__ A, const __bf16* __restrict__ Bt,
                       const float* __restrict__ bias, float* __restrict__ Cf,
                       __bf16* __restrict__ Cb, int M, int N, int Kd) {
  int wid  = (blockIdx.x * blockDim.x + threadIdx.x) >> 5;
  int lane = threadIdx.x & 31;
  int tilesN = N >> 6;
  if (wid >= (M >> 4) * tilesN) return;
  int tm = wid / tilesN, tn = wid - tm * tilesN;
  int m = lane & 15, half = lane >> 4;
  const __bf16* ar = A  + (size_t)(tm * 16 + m) * Kd;
  const __bf16* br = Bt + (size_t)(tn * 64 + m) * Kd;
  f32x8 acc[4] = {};
  for (int kd = 0; kd < Kd; kd += 32) {
    BFrag a, b;
    a.u[0] = *(const uint4*)(ar + kd + 16 * half);
    a.u[1] = *(const uint4*)(ar + kd + 16 * half + 8);
    const __bf16* bp = br + kd + 16 * half;
    #pragma unroll
    for (int t = 0; t < 4; ++t) {
      b.u[0] = *(const uint4*)(bp + (size_t)t * 16 * Kd);
      b.u[1] = *(const uint4*)(bp + (size_t)t * 16 * Kd + 8);
      acc[t] = wmma_bf16(a, b, acc[t]);
    }
  }
  #pragma unroll
  for (int t = 0; t < 4; ++t) {
    int col = tn * 64 + t * 16 + m;
    float bv = bias ? bias[col] : 0.0f;
    #pragma unroll
    for (int r = 0; r < 8; ++r) {
      int row = tm * 16 + r + 8 * half;
      float v = acc[t][r] + bv;
      if (Cb) Cb[(size_t)row * N + col] = (__bf16)v;
      else    Cf[(size_t)row * N + col] = v;
    }
  }
}

// ---------------- LDS double-buffered WMMA GEMM (big GEMMs) ----------------
// Block (256 thr / 8 waves) computes a 128x128 C tile; wave = 32x64 (8 accums).
// K staged in 32-wide tiles, async global->LDS, double buffered.
// Requires M%128==0, N%128==0, K%32==0.
static constexpr int LDT = 40;   // LDS row stride in bf16 (80B = 5*16B: aligned + bank-spread)

__global__ void __launch_bounds__(256)
k_gemm_lds(const __bf16* __restrict__ A, const __bf16* __restrict__ Bt,
           const float* __restrict__ bias, float* __restrict__ Cf,
           __bf16* __restrict__ Cb, int M, int N, int Kd) {
  __shared__ __bf16 As[2][128 * LDT];
  __shared__ __bf16 Bs[2][128 * LDT];

  const int tilesN = N >> 7;
  const int tn0 = blockIdx.x % tilesN;
  const int tm0 = blockIdx.x / tilesN;
  const int tid  = threadIdx.x;
  const int wave = tid >> 5, lane = tid & 31;
  const int wm = wave >> 1, wn = wave & 1;          // 4 x 2 wave grid
  const int m = lane & 15, half = lane >> 4;

  const __bf16* gA = A  + (size_t)(tm0 * 128) * Kd;
  const __bf16* gB = Bt + (size_t)(tn0 * 128) * Kd;

  // stage one 128x32 K-tile of A and B into LDS buffer `buf`
  auto stage = [&](int buf, int kt) {
    #pragma unroll
    for (int t = 0; t < 2; ++t) {
      int c    = tid + t * 256;          // 512 chunks of 16B per matrix
      int row  = c >> 2;
      int col8 = (c & 3) * 8;
      cp_b128(&As[buf][row * LDT + col8], gA + (size_t)row * Kd + kt * 32 + col8);
      cp_b128(&Bs[buf][row * LDT + col8], gB + (size_t)row * Kd + kt * 32 + col8);
    }
  };

  f32x8 acc[2][4] = {};
  const int KT = Kd >> 5;
  stage(0, 0);
  for (int kt = 0; kt < KT; ++kt) {
    const int buf = kt & 1;
    async_wait();
    __syncthreads();                      // tile `buf` ready; all waves done with `buf^1`
    if (kt + 1 < KT) stage(buf ^ 1, kt + 1);

    BFrag a0, a1;
    {
      const __bf16* ap0 = &As[buf][(wm * 32 + m) * LDT + 16 * half];
      const __bf16* ap1 = &As[buf][(wm * 32 + 16 + m) * LDT + 16 * half];
      a0.u[0] = *(const uint4*)ap0;  a0.u[1] = *(const uint4*)(ap0 + 8);
      a1.u[0] = *(const uint4*)ap1;  a1.u[1] = *(const uint4*)(ap1 + 8);
    }
    #pragma unroll
    for (int t = 0; t < 4; ++t) {
      BFrag b;
      const __bf16* bp = &Bs[buf][(wn * 64 + t * 16 + m) * LDT + 16 * half];
      b.u[0] = *(const uint4*)bp;  b.u[1] = *(const uint4*)(bp + 8);
      acc[0][t] = wmma_bf16(a0, b, acc[0][t]);
      acc[1][t] = wmma_bf16(a1, b, acc[1][t]);
    }
  }

  #pragma unroll
  for (int r = 0; r < 2; ++r)
    #pragma unroll
    for (int t = 0; t < 4; ++t) {
      int col = tn0 * 128 + wn * 64 + t * 16 + m;
      float bv = bias ? bias[col] : 0.0f;
      #pragma unroll
      for (int rr = 0; rr < 8; ++rr) {
        int row = tm0 * 128 + wm * 32 + r * 16 + rr + 8 * half;
        float v = acc[r][t][rr] + bv;
        if (Cb) Cb[(size_t)row * N + col] = (__bf16)v;
        else    Cf[(size_t)row * N + col] = v;
      }
    }
}

// scores tile: one wave = one 16x16 score tile of one (batch-chunk, head); K=32 single step
__global__ void k_scores(const __bf16* __restrict__ qkv, float* __restrict__ sc, int b0) {
  int wid  = (blockIdx.x * blockDim.x + threadIdx.x) >> 5;
  int lane = threadIdx.x & 31;
  if (wid >= CB * NHv * 16 * 16) return;
  int tk = wid & 15, tq = (wid >> 4) & 15, hh = (wid >> 8) & 7, bc = wid >> 11;
  int b = b0 + bc;
  int m = lane & 15, half = lane >> 4;
  BFrag a, bb;
  const __bf16* qp = qkv + (size_t)(b * Sq + tq * 16 + m) * 768 + hh * 32;
  a.u[0] = *(const uint4*)(qp + 16 * half);
  a.u[1] = *(const uint4*)(qp + 16 * half + 8);
  const __bf16* kp = qkv + (size_t)(b * Sq + tk * 16 + m) * 768 + 256 + hh * 32 + 16 * half;
  bb.u[0] = *(const uint4*)kp;
  bb.u[1] = *(const uint4*)(kp + 8);
  f32x8 acc = {};
  acc = wmma_bf16(a, bb, acc);
  const float scale = 0.17677669529663687f;   // 1/sqrt(32)
  float* out = sc + ((size_t)(bc * NHv + hh) * Sq + tq * 16) * Sq + tk * 16;
  #pragma unroll
  for (int r = 0; r < 8; ++r)
    out[(size_t)(r + 8 * half) * Sq + m] = acc[r] * scale;
}

// att@v: one wave = 16 query rows x 32 head dims of one (bc,h); probs bf16 @ vT -> o bf16
__global__ void k_attv(const __bf16* __restrict__ pr, const __bf16* __restrict__ vT,
                       __bf16* __restrict__ o, int b0) {
  int wid  = (blockIdx.x * blockDim.x + threadIdx.x) >> 5;
  int lane = threadIdx.x & 31;
  if (wid >= CB * NHv * 16) return;
  int tq = wid & 15, hh = (wid >> 4) & 7, bc = wid >> 7;
  int b = b0 + bc;
  int m = lane & 15, half = lane >> 4;
  int p = bc * NHv + hh;
  const __bf16* ar = pr + ((size_t)p * Sq + tq * 16 + m) * Sq;
  const __bf16* br = vT + (size_t)p * HD * Sq + (size_t)m * Sq;
  f32x8 acc[2] = {};
  for (int kd = 0; kd < Sq; kd += 32) {
    BFrag a, bf;
    a.u[0] = *(const uint4*)(ar + kd + 16 * half);
    a.u[1] = *(const uint4*)(ar + kd + 16 * half + 8);
    #pragma unroll
    for (int t = 0; t < 2; ++t) {
      const __bf16* bp = br + (size_t)t * 16 * Sq + kd + 16 * half;
      bf.u[0] = *(const uint4*)bp;
      bf.u[1] = *(const uint4*)(bp + 8);
      acc[t] = wmma_bf16(a, bf, acc[t]);
    }
  }
  #pragma unroll
  for (int t = 0; t < 2; ++t)
    #pragma unroll
    for (int r = 0; r < 8; ++r) {
      int row = b * Sq + tq * 16 + r + 8 * half;
      int col = hh * 32 + t * 16 + m;
      o[(size_t)row * Dv + col] = (__bf16)acc[t][r];
    }
}

// ---------------- host orchestration ----------------
extern "C" void kernel_launch(void* const* d_in, const int* in_sizes, int n_in,
                              void* d_out, int out_size, void* d_ws, size_t ws_size,
                              hipStream_t stream) {
  (void)in_sizes; (void)n_in; (void)out_size; (void)ws_size;
  const float* x          = (const float*)d_in[0];
  const float* val_w      = (const float*)d_in[1];
  const float* val_b      = (const float*)d_in[2];
  const float* id_embed   = (const float*)d_in[3];
  const float* attn_in_w  = (const float*)d_in[4];
  const float* attn_in_b  = (const float*)d_in[5];
  const float* attn_out_w = (const float*)d_in[6];
  const float* attn_out_b = (const float*)d_in[7];
  const float* norm1_w    = (const float*)d_in[8];
  const float* norm1_b    = (const float*)d_in[9];
  const float* norm2_w    = (const float*)d_in[10];
  const float* norm2_b    = (const float*)d_in[11];
  const float* ffn1_coef  = (const float*)d_in[12];
  const float* ffn1_sb    = (const float*)d_in[13];
  const float* ffn1_ssp   = (const float*)d_in[14];
  const float* ffn2_coef  = (const float*)d_in[15];
  const float* ffn2_sb    = (const float*)d_in[16];
  const float* ffn2_ssp   = (const float*)d_in[17];
  const float* fin_w      = (const float*)d_in[18];
  const float* fin_b      = (const float*)d_in[19];
  const float* h1_coef    = (const float*)d_in[20];
  const float* h1_sb      = (const float*)d_in[21];
  const float* h1_ssp     = (const float*)d_in[22];
  const float* h2_coef    = (const float*)d_in[23];
  const float* h2_sb      = (const float*)d_in[24];
  const float* h2_ssp     = (const float*)d_in[25];
  const float* reg_w      = (const float*)d_in[26];
  const float* reg_b      = (const float*)d_in[27];

  // workspace carve-up (~190 MB total)
  char* ws = (char*)d_ws;
  size_t off = 0;
  auto carve = [&](size_t bytes) -> char* {
    char* p = ws + off;
    off = (off + bytes + 255) & ~(size_t)255;
    return p;
  };
  float*  H    = (float*) carve((size_t)NTOK * Dv * 4);          // activations
  __bf16* HB   = (__bf16*)carve((size_t)NTOK * Dv * 2);
  __bf16* QKVB = (__bf16*)carve((size_t)NTOK * 768 * 2);
  float*  SC   = (float*) carve((size_t)CB * NHv * Sq * Sq * 4); // score chunk
  __bf16* PR   = (__bf16*)carve((size_t)CB * NHv * Sq * Sq * 2); // probs chunk
  __bf16* VT   = (__bf16*)carve((size_t)CB * NHv * HD * Sq * 2);
  __bf16* OB   = (__bf16*)carve((size_t)NTOK * Dv * 2);          // attention output
  float*  TMPD = (float*) carve((size_t)NTOK * Dv * 4);          // a / f2
  float*  F1   = (float*) carve((size_t)CN * Hv * 4);            // f1 chunk
  __bf16* PHI  = (__bf16*)carve((size_t)CN * Hv * 9 * 2);        // phi chunk (max 4608 wide)
  __bf16* WQKV = (__bf16*)carve((size_t)768 * 256 * 2);
  __bf16* WOUT = (__bf16*)carve((size_t)256 * 256 * 2);
  __bf16* WT1  = (__bf16*)carve((size_t)512 * 2304 * 2);
  __bf16* WT2  = (__bf16*)carve((size_t)256 * 4608 * 2);
  __bf16* WH1  = (__bf16*)carve((size_t)512 * 2304 * 2);
  __bf16* WH2  = (__bf16*)carve((size_t)512 * 4608 * 2);
  float*  POOL = (float*) carve((size_t)Bv * Dv * 4);
  float*  K1   = (float*) carve((size_t)Bv * Hv * 4);
  float*  K2   = (float*) carve((size_t)Bv * Hv * 4);

  const dim3 blk(256);
  auto nb = [](long n)     { return dim3((unsigned)((n + 255) / 256)); };
  auto wb = [](long waves) { return dim3((unsigned)((waves + 7) / 8)); };
  auto gb = [](int M, int N) { return dim3((unsigned)((M / 128) * (N / 128))); };

  // 1) embedding
  k_embed<<<nb((long)NTOK * Dv), blk, 0, stream>>>(x, val_w, val_b, id_embed, H);

  for (int l = 0; l < Lv; ++l) {
    // weight prep (bf16, already N-major for Bt since reference uses x @ W^T)
    k_f32_to_bf16<<<nb(768 * 256), blk, 0, stream>>>(attn_in_w + (size_t)l * 768 * 256, WQKV, 768 * 256);
    k_f32_to_bf16<<<nb(256 * 256), blk, 0, stream>>>(attn_out_w + (size_t)l * 256 * 256, WOUT, 256 * 256);

    // QKV projection -> bf16 (M=16384, N=768, K=256)
    k_f32_to_bf16<<<nb((long)NTOK * Dv), blk, 0, stream>>>(H, HB, NTOK * Dv);
    k_gemm_lds<<<gb(NTOK, 768), blk, 0, stream>>>(
        HB, WQKV, attn_in_b + (size_t)l * 768, nullptr, QKVB, NTOK, 768, 256);

    // attention, chunked over batches
    for (int c = 0; c < Bv / CB; ++c) {
      int b0 = c * CB;
      k_scores <<<wb((long)CB * NHv * 256), blk, 0, stream>>>(QKVB, SC, b0);
      k_softmax<<<wb((long)CB * NHv * Sq),  blk, 0, stream>>>(SC, PR, CB * NHv * Sq);
      k_vtrans <<<nb((long)CB * NHv * HD * Sq), blk, 0, stream>>>(QKVB, VT, b0);
      k_attv   <<<wb((long)CB * NHv * 16),  blk, 0, stream>>>(PR, VT, OB, b0);
    }

    // output projection + residual LN1
    k_gemm_lds<<<gb(NTOK, 256), blk, 0, stream>>>(
        OB, WOUT, attn_out_b + (size_t)l * 256, TMPD, nullptr, NTOK, 256, 256);
    k_resid_ln<<<wb(NTOK), blk, 0, stream>>>(H, TMPD, norm1_w + l * 256, norm1_b + l * 256, NTOK);

    // KAN FFN weights
    k_kan_weight<<<nb(256 * 512), blk, 0, stream>>>(
        ffn1_coef + (size_t)l * 256 * 512 * 8, ffn1_sb + (size_t)l * 256 * 512,
        ffn1_ssp + (size_t)l * 256 * 512, WT1, 256, 512);
    k_kan_weight<<<nb(512 * 256), blk, 0, stream>>>(
        ffn2_coef + (size_t)l * 512 * 256 * 8, ffn2_sb + (size_t)l * 512 * 256,
        ffn2_ssp + (size_t)l * 512 * 256, WT2, 512, 256);

    // KAN FFN, row-chunked: phi1 -> GEMM -> f1 -> phi2 -> GEMM -> f2 (into TMPD)
    for (int c = 0; c < NTOK / CN; ++c) {
      int r0 = c * CN;
      k_phi<<<nb((long)CN * 256), blk, 0, stream>>>(H + (size_t)r0 * 256, PHI, CN * 256);
      k_gemm_lds<<<gb(CN, 512), blk, 0, stream>>>(
          PHI, WT1, nullptr, F1, nullptr, CN, 512, 2304);
      k_phi<<<nb((long)CN * 512), blk, 0, stream>>>(F1, PHI, CN * 512);
      k_gemm_lds<<<gb(CN, 256), blk, 0, stream>>>(
          PHI, WT2, nullptr, TMPD + (size_t)r0 * 256, nullptr, CN, 256, 4608);
    }
    k_resid_ln<<<wb(NTOK), blk, 0, stream>>>(H, TMPD, norm2_w + l * 256, norm2_b + l * 256, NTOK);
  }

  // final LN + mean pool
  k_resid_ln<<<wb(NTOK), blk, 0, stream>>>(H, nullptr, fin_w, fin_b, NTOK);
  k_pool<<<nb(Bv * Dv), blk, 0, stream>>>(H, POOL);

  // KAN head 1: (64 x 2304) @ (2304 x 512)
  k_kan_weight<<<nb(256 * 512), blk, 0, stream>>>(h1_coef, h1_sb, h1_ssp, WH1, 256, 512);
  k_phi<<<nb(Bv * 256), blk, 0, stream>>>(POOL, PHI, Bv * 256);
  k_gemm<<<wb((long)(Bv / 16) * (512 / 64)), blk, 0, stream>>>(
      PHI, WH1, nullptr, K1, nullptr, Bv, 512, 2304);

  // KAN head 2: (64 x 4608) @ (4608 x 512)
  k_kan_weight<<<nb(512 * 512), blk, 0, stream>>>(h2_coef, h2_sb, h2_ssp, WH2, 512, 512);
  k_phi<<<nb(Bv * 512), blk, 0, stream>>>(K1, PHI, Bv * 512);
  k_gemm<<<wb((long)(Bv / 16) * (512 / 64)), blk, 0, stream>>>(
      PHI, WH2, nullptr, K2, nullptr, Bv, 512, 4608);

  // regression head
  k_reg<<<nb(Bv * OUTv), blk, 0, stream>>>(K2, reg_w, reg_b, (float*)d_out);
}
